// RCMModule_61014305407304
// MI455X (gfx1250) — matrix operand order repaired and checked by reading, not compiled, not run on hardware
//
#include <hip/hip_runtime.h>
#include <hip/hip_bf16.h>

typedef __attribute__((ext_vector_type(16))) _Float16 v16h;
typedef __attribute__((ext_vector_type(8)))  _Float16 v8h;
typedef __attribute__((ext_vector_type(8)))  float    v8f;

#define CDIM 512
#define TDIM 32
#define DH   16
#define HWN  196
#define QKW  (TDIM*DH)            // 512
#define NTILES 13                 // ceil(196/16)
#define OUT_ELEMS 25690112        // 8*512*32*196
#define K3_BLOCK 512

// ---------------- Kernel 1: spatial mean pool + f16 copy of x ----------------
__global__ __launch_bounds__(256) void k_pool(const float* __restrict__ x,
                                              float* __restrict__ pooled,
                                              _Float16* __restrict__ xh) {
  int wid  = threadIdx.x >> 5;
  int lane = threadIdx.x & 31;
  int idx  = blockIdx.x * 8 + wid;            // (b*512+c)*32+t, 131072 total
  const float* row = x + (size_t)idx * HWN;
  _Float16* hrow = xh + (size_t)idx * HWN;
  float s = 0.f;
  for (int i = lane; i < HWN; i += 32) {
    float v = row[i];
    s += v;
    hrow[i] = (_Float16)v;
  }
#pragma unroll
  for (int m = 16; m >= 1; m >>= 1) s += __shfl_xor(s, m, 32);
  if (lane == 0) pooled[idx] = s * (1.0f / HWN);
}

// ---------------- Kernel 2: q/k projections -> f16 (scale folded into q) ----------------
__global__ __launch_bounds__(512) void k_qk(const float* __restrict__ pooled,
                                            const float* __restrict__ Wq,
                                            const float* __restrict__ Wk,
                                            _Float16* __restrict__ qh,
                                            _Float16* __restrict__ kh) {
  __shared__ float s_p[TDIM];
  int bc = blockIdx.x;          // 0..4095 = b*512+c
  int o  = threadIdx.x;         // 0..511
  if (o < TDIM) s_p[o] = pooled[bc * TDIM + o];
  __syncthreads();
  float aq = 0.f, ak = 0.f;
#pragma unroll
  for (int t = 0; t < TDIM; ++t) {
    float p = s_p[t];
    aq = fmaf(p, Wq[t * QKW + o], aq);
    ak = fmaf(p, Wk[t * QKW + o], ak);
  }
  size_t oi = (size_t)bc * QKW + o;
  qh[oi] = (_Float16)(aq * 0.25f);   // fold DH^-0.5 into q
  kh[oi] = (_Float16)ak;
}

// ---------------- Kernel 3: attention + attn@v via WMMA ----------------
// One block per (b, h, iblk); 16 waves, each owns a 16-row M-tile of c.
__global__ __launch_bounds__(K3_BLOCK) void k_attn(
    const _Float16* __restrict__ xh, const _Float16* __restrict__ qh,
    const _Float16* __restrict__ kh, float* __restrict__ out,
    float* __restrict__ te) {
  __shared__ __align__(16) _Float16 vT[NTILES * 16][40];   // [col][j], 16640 B
  __shared__ __align__(16) _Float16 stg[16][16][40];       // wave-private transpose, 20480 B
  __shared__ __align__(16) _Float16 s_kh[CDIM][24];        // k rows (16 valid + pad), 24576 B
  __shared__ float s_te;

  int bh   = blockIdx.x >> 1;
  int iblk = blockIdx.x & 1;
  int b = bh >> 5;
  int h = bh & 31;
  int tid  = threadIdx.x;
  int wid  = tid >> 5;
  int lane = tid & 31;
  int lh   = lane & 15;
  bool hi  = lane >= 16;
  if (tid == 0) s_te = 0.f;

  int rowbase = iblk * 256 + wid * 16;

  // ---- async-copy all 512 k rows (16 f16 each) for this (b,h) into LDS ----
  {
    unsigned ldsbase = (unsigned)(size_t)(&s_kh[0][0]);
    unsigned long long gbase =
        (unsigned long long)(size_t)(kh + ((size_t)b * CDIM) * QKW + h * DH);
#pragma unroll
    for (int c = 0; c < 2; ++c) {
      int chunk = tid + c * K3_BLOCK;            // 0..1023 -> 1024 x 16B
      int row  = chunk >> 1;
      int half = chunk & 1;
      unsigned lofs = ldsbase + row * 48 + half * 16;
      unsigned goff = (unsigned)(row * (QKW * 2) + half * 16);  // bytes
      asm volatile("global_load_async_to_lds_b128 %0, %1, %2 offset:0"
                   :: "v"(lofs), "v"(goff), "s"(gbase) : "memory");
    }
    asm volatile("s_wait_asynccnt 0" ::: "memory");
  }

  // A fragment of q (16x32, K=16 valid, upper K zero-padded); scale pre-folded
  v16h aq;
  {
    const _Float16* qp = qh + ((size_t)b * CDIM + (rowbase + lh)) * QKW + h * DH;
    v8h qv0 = *(const v8h*)qp;
    v8h qv1 = *(const v8h*)(qp + 8);
#pragma unroll
    for (int e = 0; e < 8; ++e) { aq[e] = hi ? (_Float16)0.f : qv0[e]; aq[e + 8] = hi ? (_Float16)0.f : qv1[e]; }
  }

  v8f acc[NTILES];
#pragma unroll
  for (int n = 0; n < NTILES; ++n) acc[n] = (v8f){};
  float sume[8];
#pragma unroll
  for (int r = 0; r < 8; ++r) sume[r] = 0.f;

  // software pipeline: prefetch v tile (raw u16) into registers
  const unsigned short* xhu = (const unsigned short*)xh;
  unsigned short vreg[NTILES];
#pragma unroll
  for (int t = 0; t < NTILES; ++t) {                 // 32*208 == 13*512 exactly
    int i = tid + t * K3_BLOCK;
    int j = i / (NTILES * 16), col = i - j * (NTILES * 16);
    vreg[t] = (col < HWN)
        ? xhu[(((size_t)b * CDIM + j) * TDIM + h) * HWN + col] : (unsigned short)0;
  }

  for (int jb = 0; jb < CDIM; jb += 32) {
    __syncthreads();
#pragma unroll
    for (int t = 0; t < NTILES; ++t) {
      int i = tid + t * K3_BLOCK;
      int j = i / (NTILES * 16), col = i - j * (NTILES * 16);
      *(unsigned short*)&vT[col][j] = vreg[t];
    }
    __syncthreads();

    // prefetch next tile while this iteration computes
    if (jb + 32 < CDIM) {
#pragma unroll
      for (int t = 0; t < NTILES; ++t) {
        int i = tid + t * K3_BLOCK;
        int j = i / (NTILES * 16), col = i - j * (NTILES * 16);
        vreg[t] = (col < HWN)
            ? xhu[(((size_t)b * CDIM + (jb + 32 + j)) * TDIM + h) * HWN + col]
            : (unsigned short)0;
      }
    }

    // B fragments of k^T from LDS (lanes>=16 = K 16..31 padding -> zero)
    v16h bk0, bk1;
    if (!hi) {
      const v8h* k0 = (const v8h*)&s_kh[jb + lh][0];
      const v8h* k1 = (const v8h*)&s_kh[jb + 16 + lh][0];
      bk0 = __builtin_shufflevector(k0[0], k0[1], 0, 1, 2, 3, 4, 5, 6, 7,
                                    8, 9, 10, 11, 12, 13, 14, 15);
      bk1 = __builtin_shufflevector(k1[0], k1[1], 0, 1, 2, 3, 4, 5, 6, 7,
                                    8, 9, 10, 11, 12, 13, 14, 15);
    } else {
#pragma unroll
      for (int e = 0; e < 16; ++e) { bk0[e] = (_Float16)0.f; bk1[e] = (_Float16)0.f; }
    }
    v8f z = (v8f){};
    v8f d0 = __builtin_amdgcn_wmma_f32_16x16x32_f16(false, aq, false, bk0, (short)0, z, false, false);
    v8f d1 = __builtin_amdgcn_wmma_f32_16x16x32_f16(false, aq, false, bk1, (short)0, z, false, false);

    // exp (f32) + stage to LDS (wave-private; same-wave DS ops are in order)
#pragma unroll
    for (int r = 0; r < 8; ++r) {
      float e0 = __expf(d0[r]);
      float e1 = __expf(d1[r]);
      sume[r] += e0 + e1;
      int row = hi ? (r + 8) : r;
      stg[wid][row][lh]      = (_Float16)e0;
      stg[wid][row][lh + 16] = (_Float16)e1;
    }

    // read attn back in A-matrix layout (transpose via LDS)
    int c0 = hi ? 8 : 0;
    int c1 = hi ? 24 : 16;
    v8h a0 = *(const v8h*)&stg[wid][lh][c0];
    v8h a1 = *(const v8h*)&stg[wid][lh][c1];
    v16h aa = __builtin_shufflevector(a0, a1, 0, 1, 2, 3, 4, 5, 6, 7,
                                      8, 9, 10, 11, 12, 13, 14, 15);

    // out(16x208) += attn(16x32) @ v(32x208)
#pragma unroll
    for (int n = 0; n < NTILES; ++n) {
      int col  = n * 16 + lh;
      int kofs = hi ? 8 : 0;
      v8h blo = *(const v8h*)&vT[col][kofs];
      v8h bhi = *(const v8h*)&vT[col][kofs + 16];
      v16h bv = __builtin_shufflevector(blo, bhi, 0, 1, 2, 3, 4, 5, 6, 7,
                                        8, 9, 10, 11, 12, 13, 14, 15);
      acc[n] = __builtin_amdgcn_wmma_f32_16x16x32_f16(false, aa, false, bv, (short)0,
                                                      acc[n], false, false);
    }
  }

  // softmax denominators: reduce across the 16 column-lanes of each half
  float recip[8];
#pragma unroll
  for (int r = 0; r < 8; ++r) {
    float s = sume[r];
    s += __shfl_xor(s, 1, 16);
    s += __shfl_xor(s, 2, 16);
    s += __shfl_xor(s, 4, 16);
    s += __shfl_xor(s, 8, 16);
    recip[r] = 1.0f / s;
  }

  // normalize, store, and accumulate te contribution
  float tacc = 0.f;
#pragma unroll
  for (int n = 0; n < NTILES; ++n) {
    int col = n * 16 + lh;
    if (col < HWN) {
#pragma unroll
      for (int r = 0; r < 8; ++r) {
        int row = rowbase + r + (hi ? 8 : 0);
        float v = acc[n][r] * recip[r];
        out[(((size_t)b * CDIM + row) * TDIM + h) * HWN + col] = v;
        tacc += v;
      }
    }
  }
#pragma unroll
  for (int m = 16; m >= 1; m >>= 1) tacc += __shfl_xor(tacc, m, 32);
  if (lane == 0) atomicAdd(&s_te, tacc);
  __syncthreads();
  if (tid == 0) atomicAdd(&te[b * TDIM + h], s_te * (1.0f / ((float)CDIM * HWN)));
}

// ---------------- Kernel 4: distill MLP + LayerNorm + sigmoid ----------------
__global__ __launch_bounds__(256) void k_mlp(const float* __restrict__ te,
    const float* __restrict__ W1, const float* __restrict__ W2,
    const float* __restrict__ ln_g, const float* __restrict__ ln_b,
    float* __restrict__ tpn, float* __restrict__ tw) {
  __shared__ float s_te[8][TDIM];
  __shared__ float s_h[8][64];
  __shared__ float s_tp[8][TDIM];
  int tid = threadIdx.x;
  s_te[tid >> 5][tid & 31] = te[tid];
  __syncthreads();
  for (int i = tid; i < 8 * 64; i += 256) {
    int bb = i >> 6, j = i & 63;
    float a = 0.f;
#pragma unroll
    for (int t = 0; t < TDIM; ++t) a = fmaf(s_te[bb][t], W1[t * 64 + j], a);
    s_h[bb][j] = 0.5f * a * (1.0f + erff(a * 0.70710678118654752f));  // exact gelu
  }
  __syncthreads();
  {
    int bb = tid >> 5, t = tid & 31;
    float a = 0.f;
#pragma unroll
    for (int j = 0; j < 64; ++j) a = fmaf(s_h[bb][j], W2[j * TDIM + t], a);
    s_tp[bb][t] = a;
  }
  __syncthreads();
  {
    int bb = tid >> 5, t = tid & 31;
    float mu = 0.f;
#pragma unroll
    for (int j = 0; j < TDIM; ++j) mu += s_tp[bb][j];
    mu *= (1.0f / TDIM);
    float var = 0.f;
#pragma unroll
    for (int j = 0; j < TDIM; ++j) { float d = s_tp[bb][j] - mu; var = fmaf(d, d, var); }
    var *= (1.0f / TDIM);
    float v = (s_tp[bb][t] - mu) * rsqrtf(var + 1e-5f) * ln_g[t] + ln_b[t];
    tpn[tid] = v;
    tw[tid]  = 1.0f / (1.0f + __expf(-v));
  }
}

// ---------------- Kernel 5: logits = tpn @ Wl + bl ----------------
__global__ __launch_bounds__(256) void k_logits(const float* __restrict__ tpn,
    const float* __restrict__ Wl, const float* __restrict__ bl,
    float* __restrict__ logits) {
  int gid = blockIdx.x * 256 + threadIdx.x;  // 4096
  int bb = gid >> 9, o = gid & 511;
  float a = bl[o];
#pragma unroll
  for (int t = 0; t < TDIM; ++t) a = fmaf(tpn[bb * TDIM + t], Wl[t * 512 + o], a);
  logits[gid] = a;
}

// ---------------- Kernel 6: out = out*tw + residual (float4) ----------------
__global__ __launch_bounds__(256) void k_final(const float* __restrict__ x,
    const float* __restrict__ tw, float* __restrict__ out) {
  size_t gid = (size_t)blockIdx.x * 256 + threadIdx.x;  // OUT_ELEMS/4
  size_t idx = gid * 4;
  int t  = (int)((idx / HWN) & 31);
  int bb = (int)(idx / ((size_t)CDIM * TDIM * HWN));
  float w = tw[bb * TDIM + t];
  const float4 xi = ((const float4*)x)[gid];
  float4 o = ((float4*)out)[gid];
  o.x = fmaf(o.x, w, xi.x);
  o.y = fmaf(o.y, w, xi.y);
  o.z = fmaf(o.z, w, xi.z);
  o.w = fmaf(o.w, w, xi.w);
  ((float4*)out)[gid] = o;
}

extern "C" void kernel_launch(void* const* d_in, const int* in_sizes, int n_in,
                              void* d_out, int out_size, void* d_ws, size_t ws_size,
                              hipStream_t stream) {
  (void)in_sizes; (void)n_in; (void)out_size; (void)ws_size;
  const float* x   = (const float*)d_in[0];
  const float* Wq  = (const float*)d_in[1];
  const float* Wk  = (const float*)d_in[2];
  const float* W1  = (const float*)d_in[3];
  const float* W2  = (const float*)d_in[4];
  const float* lng = (const float*)d_in[5];
  const float* lnb = (const float*)d_in[6];
  const float* Wl  = (const float*)d_in[7];
  const float* bl  = (const float*)d_in[8];
  float* out    = (float*)d_out;
  float* logits = out + OUT_ELEMS;

  // workspace layout
  _Float16* xh = (_Float16*)d_ws;                 // 25,690,112 halfs
  _Float16* qh = xh + OUT_ELEMS;                  // 2,097,152 halfs
  _Float16* kh = qh + (size_t)8 * CDIM * QKW;     // 2,097,152 halfs
  float* pooled = (float*)(kh + (size_t)8 * CDIM * QKW);   // 131072 f32 (offset is 4B-aligned)
  float* te  = pooled + 131072;                   // 256 f32
  float* tpn = te + 256;                          // 256 f32
  float* tw  = tpn + 256;                         // 256 f32

  hipMemsetAsync(te, 0, 256 * sizeof(float), stream);
  k_pool  <<<16384, 256, 0, stream>>>(x, pooled, xh);
  k_qk    <<<4096, 512, 0, stream>>>(pooled, Wq, Wk, qh, kh);
  k_attn  <<<512, K3_BLOCK, 0, stream>>>(xh, qh, kh, out, te);
  k_mlp   <<<1, 256, 0, stream>>>(te, W1, W2, lng, lnb, tpn, tw);
  k_logits<<<16, 256, 0, stream>>>(tpn, Wl, bl, logits);
  k_final <<<25088, 256, 0, stream>>>(x, tw, out);
}
